// RGCNLowMem_4475355922763
// MI455X (gfx1250) — matrix operand order repaired and checked by reading, not compiled, not run on hardware
//
#include <hip/hip_runtime.h>

// RGCN low-mem forward, fused edge-centric WMMA kernel for gfx1250 (MI455X).
//
//  out[v] = sum_{e: dst[e]==v} feat[src[e]] @ weight[etype[e]]
//
// feat (25.6MB) + out (25.6MB) + weights fit in the 192MB L2, so a fused
// gather -> WMMA -> atomic-scatter kernel is bandwidth-optimal. Each block
// counting-sorts 512 edges by relation in LDS, then each wave runs 16-edge
// chunks through 8x v_wmma_f32_16x16x32_bf16 (4 N-tiles x 2 K-chunks) and
// scatters with unconditional f32 atomics (padding rows redirect to a dump
// row in scratch -> no exec-mask branching in the epilogue).

#define BLOCK   256
#define TILE    512
#define NWAVES  (BLOCK / 32)
#define EPT     (TILE / BLOCK)   // edges per thread during bucketing

typedef __attribute__((ext_vector_type(16))) __bf16 v16bf;
typedef __attribute__((ext_vector_type(8)))  float  v8f;

union BF16x16 {
  v16bf v;
  unsigned short u[16];
  uint4 q[2];
};

__device__ __forceinline__ unsigned short f2bf(float f) {
  unsigned int u = __float_as_uint(f);
  u += 0x7FFFu + ((u >> 16) & 1u);   // round-to-nearest-even
  return (unsigned short)(u >> 16);
}

// ---------------------------------------------------------------- zero output
__global__ void rgcn_zero_f32(float* __restrict__ p, int n) {
  int i = blockIdx.x * blockDim.x + threadIdx.x;
  int s = gridDim.x * blockDim.x;
  for (; i < n; i += s) p[i] = 0.0f;
}

// ------------------------------------------- weight f32 -> bf16, transposed
// in : w [16][64 (k)][64 (n)]  f32
// out: wt[16][64 (n)][64 (k)]  bf16   (B-fragment rows become contiguous)
__global__ void rgcn_wconv(const float* __restrict__ w,
                           unsigned short* __restrict__ wt) {
  int i = blockIdx.x * blockDim.x + threadIdx.x;  // 0 .. 16*64*64-1
  int r = i >> 12;
  int k = (i >> 6) & 63;
  int n = i & 63;
  wt[((r << 6) + n) * 64 + k] = f2bf(w[i]);
}

// ------------------------------------------------- feat f32 -> bf16 (same
// row-major layout; A-fragment runs stay contiguous, loads become 2x b128)
__global__ void rgcn_fconv(const float* __restrict__ f,
                           unsigned short* __restrict__ fb, int n) {
  int i = blockIdx.x * blockDim.x + threadIdx.x;
  int s = gridDim.x * blockDim.x;
  for (; i < n; i += s) fb[i] = f2bf(f[i]);
}

// ------------------------------------------------------------- main kernel
template <bool PRECONV>
__global__ __launch_bounds__(BLOCK) void rgcn_edge_wmma(
    const float* __restrict__ feat,              // [N, 64] f32
    const unsigned short* __restrict__ featbf,   // [N, 64] bf16 (PRECONV)
    const unsigned short* __restrict__ wbf,      // [16][64 n][64 k] bf16
    const int* __restrict__ src,
    const int* __restrict__ dst,
    const int* __restrict__ etypes,
    float* __restrict__ out,                     // [N, 64] f32
    float* __restrict__ dump,                    // >=64 f32 scratch
    int nEdges) {
  __shared__ int s_cnt[16];
  __shared__ int s_cur[16];
  __shared__ int s_sorted[TILE];   // global edge ids, bucketed by relation
  __shared__ int s_jobs[64];       // (rel<<20)|(start<<8)|cnt, cnt<=16
  __shared__ int s_njobs;

  const int t        = threadIdx.x;
  const int tileBase = blockIdx.x * TILE;

  if (t < 16) s_cnt[t] = 0;
  __syncthreads();

  // ---- pass 1: histogram edges by relation ----
  int myE[EPT], myR[EPT];
#pragma unroll
  for (int i = 0; i < EPT; ++i) {
    int e = tileBase + t + i * BLOCK;
    if (e < nEdges) {
      int r = etypes[e];
      myE[i] = e;
      myR[i] = r;
      atomicAdd(&s_cnt[r], 1);
    } else {
      myR[i] = -1;
    }
  }
  __syncthreads();

  // ---- prefix sum + job list (serial over 16 rels, cheap) ----
  if (t == 0) {
    int acc = 0, nj = 0;
    for (int r = 0; r < 16; ++r) {
      int c = s_cnt[r];
      s_cur[r] = acc;
      int st = acc;
      acc += c;
      while (c > 0) {
        int take = c < 16 ? c : 16;
        s_jobs[nj++] = (r << 20) | (st << 8) | take;
        st += take;
        c -= take;
      }
    }
    s_njobs = nj;
  }
  __syncthreads();

  // ---- pass 2: scatter edge ids into buckets ----
#pragma unroll
  for (int i = 0; i < EPT; ++i) {
    if (myR[i] >= 0) {
      int pos = atomicAdd(&s_cur[myR[i]], 1);
      s_sorted[pos] = myE[i];
    }
  }
  __syncthreads();

  const int  wave = t >> 5;
  const int  lane = t & 31;
  const int  Mlo  = lane & 15;     // matrix row / output column sub-index
  const bool hi   = lane >= 16;
  const int  nj   = s_njobs;

  // wave-uniform job loop -> WMMA always runs with full EXEC
  for (int j = wave; j < nj; j += NWAVES) {
    const int job   = s_jobs[j];
    const int r     = job >> 20;
    const int start = (job >> 8) & 0xFFF;
    const int cnt   = job & 0xFF;

    // clamp padding rows onto a real edge; their (garbage) results are
    // redirected to `dump` in the epilogue, so no zero-fill branch needed
    const int eM   = s_sorted[start + (Mlo < cnt ? Mlo : cnt - 1)];
    const int srow = src[eM];
    const int drow = dst[eM];

    // ---- A fragments: 16x32 bf16 per K-chunk, ISA VGPR layout ----
    // lanes 0-15 (row M=lane):    k = kc+0..7  and kc+16..23
    // lanes 16-31 (row M=lane-16): k = kc+8..15 and kc+24..31
    BF16x16 A[2];
    if (PRECONV) {
      const unsigned short* fb = featbf + (size_t)srow * 64;
#pragma unroll
      for (int c = 0; c < 2; ++c) {
        const int kbase = c * 32 + (hi ? 8 : 0);
        A[c].q[0] = *(const uint4*)(fb + kbase);
        A[c].q[1] = *(const uint4*)(fb + kbase + 16);
      }
    } else {
      const float* fr = feat + (size_t)srow * 64;
#pragma unroll
      for (int c = 0; c < 2; ++c) {
        const int kbase = c * 32 + (hi ? 8 : 0);
        const float4 p0 = *(const float4*)(fr + kbase);
        const float4 p1 = *(const float4*)(fr + kbase + 4);
        const float4 p2 = *(const float4*)(fr + kbase + 16);
        const float4 p3 = *(const float4*)(fr + kbase + 20);
        A[c].u[0]  = f2bf(p0.x); A[c].u[1]  = f2bf(p0.y);
        A[c].u[2]  = f2bf(p0.z); A[c].u[3]  = f2bf(p0.w);
        A[c].u[4]  = f2bf(p1.x); A[c].u[5]  = f2bf(p1.y);
        A[c].u[6]  = f2bf(p1.z); A[c].u[7]  = f2bf(p1.w);
        A[c].u[8]  = f2bf(p2.x); A[c].u[9]  = f2bf(p2.y);
        A[c].u[10] = f2bf(p2.z); A[c].u[11] = f2bf(p2.w);
        A[c].u[12] = f2bf(p3.x); A[c].u[13] = f2bf(p3.y);
        A[c].u[14] = f2bf(p3.z); A[c].u[15] = f2bf(p3.w);
      }
    }

    v8f acc[4] = {};  // 4 N-tiles of the 16x64 chunk output, f32 accum

    // ---- B fragments from transposed bf16 weights + 8x WMMA ----
    // B (32x16): lane col = lane&15; lanes 0-15 hold k kc+0..7 and kc+16..23,
    //            lanes 16-31 hold k kc+8..15 and kc+24..31 (wt[r][n][k])
#pragma unroll
    for (int tn = 0; tn < 4; ++tn) {
      const int col = tn * 16 + Mlo;
      const unsigned short* wrow = wbf + (((r << 6) + col) << 6);
#pragma unroll
      for (int c = 0; c < 2; ++c) {
        const int kb = c * 32 + (hi ? 8 : 0);
        BF16x16 B;
        B.q[0] = *(const uint4*)(wrow + kb);
        B.q[1] = *(const uint4*)(wrow + kb + 16);
        acc[tn] = __builtin_amdgcn_wmma_f32_16x16x32_bf16(
            false, A[c].v, false, B.v, (short)0, acc[tn], false, false);
      }
    }

    // ---- scatter D (16x64) into out with unconditional f32 atomics ----
    // C/D layout: VGPR g, lane L -> row M = g + (L>=16 ? 8 : 0), col = L&15.
    // 8 hoisted lane-shuffles fetch dst rows; padding rows -> dump.
    float* base[8];
#pragma unroll
    for (int g = 0; g < 8; ++g) {
      const int Mp = g + (hi ? 8 : 0);
      const int dr = __shfl(drow, Mp, 32);   // lane Mp holds dst of row Mp
      base[g] = ((Mp < cnt) ? (out + (size_t)dr * 64) : dump) + Mlo;
    }
#pragma unroll
    for (int g = 0; g < 8; ++g) {
#pragma unroll
      for (int tn = 0; tn < 4; ++tn) {
        atomicAdd(base[g] + tn * 16, acc[tn][g]);  // imm offsets 0/64/128/192B
      }
    }
  }
}

// ---------------------------------------------------------------------------
extern "C" void kernel_launch(void* const* d_in, const int* in_sizes, int n_in,
                              void* d_out, int out_size, void* d_ws,
                              size_t ws_size, hipStream_t stream) {
  const float* feat   = (const float*)d_in[0];   // [100000, 64] f32
  const float* weight = (const float*)d_in[1];   // [16, 64, 64] f32
  const int*   src    = (const int*)d_in[2];     // [E]
  const int*   dst    = (const int*)d_in[3];     // [E]
  const int*   etyp   = (const int*)d_in[4];     // [E]
  float*       out    = (float*)d_out;           // [100000, 64] f32

  const int nEdges    = in_sizes[2];
  const int featElems = in_sizes[0];             // 100000*64

  // d_ws layout: [0,128KB) bf16 weights | [128KB,+256B) dump | bf16 feat
  char* ws = (char*)d_ws;
  unsigned short* wbf  = (unsigned short*)ws;
  float*          dump = (float*)(ws + (16 * 64 * 64 * 2));
  unsigned short* fbf  = (unsigned short*)(ws + (16 * 64 * 64 * 2) + 256);
  const size_t needPre = (size_t)(16 * 64 * 64 * 2) + 256 +
                         (size_t)featElems * sizeof(unsigned short);
  const bool preconv = ws_size >= needPre;

  // 1) zero the accumulation target
  rgcn_zero_f32<<<1024, 256, 0, stream>>>(out, out_size);

  // 2) weights f32 -> bf16, transposed for contiguous B-fragment loads
  rgcn_wconv<<<(16 * 64 * 64) / 256, 256, 0, stream>>>(weight, wbf);

  const int nTiles = (nEdges + TILE - 1) / TILE;
  if (preconv) {
    // 2b) feat f32 -> bf16 once (L2-resident; removes per-chunk VALU cvts)
    rgcn_fconv<<<1024, 256, 0, stream>>>(feat, fbf, featElems);
    rgcn_edge_wmma<true><<<nTiles, BLOCK, 0, stream>>>(
        feat, fbf, wbf, src, dst, etyp, out, dump, nEdges);
  } else {
    rgcn_edge_wmma<false><<<nTiles, BLOCK, 0, stream>>>(
        feat, fbf, wbf, src, dst, etyp, out, dump, nEdges);
  }
}
